// CGRUCell_59897613910787
// MI455X (gfx1250) — matrix operand order repaired and checked by reading, not compiled, not run on hardware
//
#include <hip/hip_runtime.h>
#include <math.h>

// Problem sizes (compile-time)
constexpr int B   = 256;
constexpr int L   = 128;
constexpr int H   = 1024;
constexpr int IN  = 1024;
constexpr int CTX = 2 * H; // 2048

typedef __attribute__((ext_vector_type(16))) __bf16        v16bf;
typedef __attribute__((ext_vector_type(2)))  __bf16        v2bf;
typedef __attribute__((ext_vector_type(8)))  float         v8f;
typedef __attribute__((ext_vector_type(8)))  unsigned int  v8u;
typedef __attribute__((ext_vector_type(4)))  unsigned int  v4u;

union V16 { v8u u; v16bf b; };

// Pack two f32 into one dword of bf16 (hardware cvt when available).
__device__ __forceinline__ unsigned int pack2_bf16(float a, float b) {
#if __has_builtin(__builtin_amdgcn_cvt_pk_bf16_f32)
  union { v2bf h; unsigned int u; } c;
  c.h = __builtin_amdgcn_cvt_pk_bf16_f32(a, b);
  return c.u;
#else
  union { __bf16 h[2]; unsigned int u; } c;
  c.h[0] = (__bf16)a;
  c.h[1] = (__bf16)b;
  return c.u;
#endif
}

// ---- bf16 operand loads (pre-converted buffers, pure b128 loads) ----
// A tile 16x32: lane (m = lane&15, hi = lane>>4) holds K runs {hi*8..+7} and
// {16+hi*8..+7}. p must point at row_base + kbase + hi*8.
__device__ __forceinline__ v16bf load_a_bf16(const __bf16* __restrict__ p) {
  V16 t;
  const v4u lo = *(const v4u*)(p);
  const v4u hi = *(const v4u*)(p + 16);
  t.u[0] = lo[0]; t.u[1] = lo[1]; t.u[2] = lo[2]; t.u[3] = lo[3];
  t.u[4] = hi[0]; t.u[5] = hi[1]; t.u[6] = hi[2]; t.u[7] = hi[3];
  return t.b;
}
// B tile 32x16: lane (n = lane&15, hi) holds K runs {hi*8..+7} and
// {16+hi*8..+7}. p must point at col_row_base + kbase + hi*8.
__device__ __forceinline__ v16bf load_b_bf16(const __bf16* __restrict__ p) {
  V16 t;
  const v4u lo = *(const v4u*)(p);
  const v4u hi = *(const v4u*)(p + 16);
  t.u[0] = lo[0]; t.u[1] = lo[1]; t.u[2] = lo[2]; t.u[3] = lo[3];
  t.u[4] = hi[0]; t.u[5] = hi[1]; t.u[6] = hi[2]; t.u[7] = hi[3];
  return t.b;
}
// A tile from an f32 source with on-the-fly hardware cvt (context only).
__device__ __forceinline__ v16bf cvt_a_f32(const float* __restrict__ p) {
  V16 t;
#pragma unroll
  for (int i = 0; i < 4; ++i) {
    t.u[i]     = pack2_bf16(p[2 * i],      p[2 * i + 1]);
    t.u[4 + i] = pack2_bf16(p[16 + 2 * i], p[16 + 2 * i + 1]);
  }
  return t.b;
}

__device__ __forceinline__ v8f wmma_bf16(v16bf a, v16bf b, v8f c) {
  return __builtin_amdgcn_wmma_f32_16x16x32_bf16(
      false, a, false, b, (short)0, c, false, false);
}

__device__ __forceinline__ float sigm(float x) {
  return 1.0f / (1.0f + __expf(-x));
}
// Fast tanh: inputs here are small (0.02-scaled weights); clamp keeps exp finite.
__device__ __forceinline__ float fast_tanh(float x) {
  const float xc = fminf(fmaxf(x, -15.0f), 15.0f);
  const float e  = __expf(2.0f * xc);
  return 1.0f - 2.0f / (e + 1.0f);
}

// ---------------------------------------------------------------------------
// Fused attention (one block of 8 waves per batch row):
//   logits = v . tanh(q + bk + context @ wk^T) + bv ; softmax ; weighted sum.
// A (context) converted on the fly with hw cvt; B (wk) pre-converted bf16.
// N tiles processed 8 at a time: per k-step = 1 A-cvt + 8 B loads + 8 WMMAs.
// (Defined first in the file so the disasm snippet shows this inner loop.)
// ---------------------------------------------------------------------------
__global__ __launch_bounds__(256) void attn_kernel(
    const float*  __restrict__ context,  // [B, L, CTX] f32
    const __bf16* __restrict__ wk,       // [H, CTX] bf16
    const float* __restrict__ bk,
    const float* __restrict__ q,         // [B, H]
    const float* __restrict__ wv,        // [1, H]
    const float* __restrict__ bv,        // [1]
    float* __restrict__ attn_out,        // [B, CTX]
    __bf16* __restrict__ attn_bf)        // [B, CTX]
{
  __shared__ float s_log[L];
  __shared__ float s_tmp[L];

  const int b    = blockIdx.x;
  const int tid  = threadIdx.x;
  const int wave = tid >> 5;
  const int lane = tid & 31;
  const int m16  = lane & 15;
  const int hi   = lane >> 4;
  const int mb   = wave * 16; // 8 waves cover L = 128 rows

  float plog[8] = {0.f, 0.f, 0.f, 0.f, 0.f, 0.f, 0.f, 0.f};

  const float* crow = context + (size_t)((size_t)b * L + mb + m16) * CTX + hi * 8;

  for (int nt0 = 0; nt0 < H / 16; nt0 += 8) {
    v8f acc[8] = {v8f{}, v8f{}, v8f{}, v8f{}, v8f{}, v8f{}, v8f{}, v8f{}};
    const __bf16* wbase = wk + (size_t)(nt0 * 16 + m16) * CTX + hi * 8;
    for (int k = 0; k < CTX; k += 32) {
      const v16bf a = cvt_a_f32(crow + k);
#pragma unroll
      for (int j = 0; j < 8; ++j)
        acc[j] = wmma_bf16(a, load_b_bf16(wbase + (size_t)j * 16 * CTX + k),
                           acc[j]);
    }
#pragma unroll
    for (int j = 0; j < 8; ++j) {
      const int ngl = (nt0 + j) * 16 + m16;
      const float qv  = q[(size_t)b * H + ngl] + bk[ngl];
      const float wvv = wv[ngl];
#pragma unroll
      for (int r = 0; r < 8; ++r)
        plog[r] += fast_tanh(acc[j][r] + qv) * wvv;
    }
  }

  // Reduce each row's partial across the 16 lanes of its half-wave.
#pragma unroll
  for (int r = 0; r < 8; ++r) {
    float v = plog[r];
    v += __shfl_xor(v, 1, 32);
    v += __shfl_xor(v, 2, 32);
    v += __shfl_xor(v, 4, 32);
    v += __shfl_xor(v, 8, 32);
    plog[r] = v;
  }
  if (m16 == 0) {
    const float bv0 = bv[0];
#pragma unroll
    for (int r = 0; r < 8; ++r)
      s_log[mb + hi * 8 + r] = plog[r] + bv0;
  }
  __syncthreads();

  // Softmax over L = 128 logits in LDS.
  if (tid < L) s_tmp[tid] = s_log[tid];
  __syncthreads();
  for (int s = L / 2; s > 0; s >>= 1) {
    if (tid < s) s_tmp[tid] = fmaxf(s_tmp[tid], s_tmp[tid + s]);
    __syncthreads();
  }
  const float mx = s_tmp[0];
  __syncthreads();
  if (tid < L) {
    const float e = __expf(s_log[tid] - mx);
    s_log[tid] = e;
    s_tmp[tid] = e;
  }
  __syncthreads();
  for (int s = L / 2; s > 0; s >>= 1) {
    if (tid < s) s_tmp[tid] += s_tmp[tid + s];
    __syncthreads();
  }
  const float inv = 1.0f / s_tmp[0];

  // Weighted context sum: thread covers 8 contiguous channels (L2-hot pass).
  const int c0 = tid * 8; // 256 * 8 = 2048 = CTX
  float a0 = 0.f, a1 = 0.f, a2 = 0.f, a3 = 0.f;
  float a4 = 0.f, a5 = 0.f, a6 = 0.f, a7 = 0.f;
  for (int l = 0; l < L; ++l) {
    const float w = s_log[l];
    const float4* cp =
        (const float4*)(context + (size_t)((size_t)b * L + l) * CTX + c0);
    const float4 u = cp[0];
    const float4 v = cp[1];
    a0 += w * u.x; a1 += w * u.y; a2 += w * u.z; a3 += w * u.w;
    a4 += w * v.x; a5 += w * v.y; a6 += w * v.z; a7 += w * v.w;
  }
  a0 *= inv; a1 *= inv; a2 *= inv; a3 *= inv;
  a4 *= inv; a5 *= inv; a6 *= inv; a7 *= inv;

  float* op = attn_out + (size_t)b * CTX + c0;
  op[0] = a0; op[1] = a1; op[2] = a2; op[3] = a3;
  op[4] = a4; op[5] = a5; op[6] = a6; op[7] = a7;

  v4u o;
  o[0] = pack2_bf16(a0, a1);
  o[1] = pack2_bf16(a2, a3);
  o[2] = pack2_bf16(a4, a5);
  o[3] = pack2_bf16(a6, a7);
  *(v4u*)(attn_bf + (size_t)b * CTX + c0) = o;
}

// ---------------------------------------------------------------------------
// f32 -> bf16 bulk conversion (one pass per reused operand).
// ---------------------------------------------------------------------------
__global__ __launch_bounds__(256) void cvt_kernel(
    const float* __restrict__ in, __bf16* __restrict__ out, unsigned int n) {
  const unsigned int i = (blockIdx.x * 256u + threadIdx.x) * 8u;
  if (i >= n) return;
  const float4* p = (const float4*)(in + i);
  const float4 a = p[0];
  const float4 b = p[1];
  v4u o;
  o[0] = pack2_bf16(a.x, a.y);
  o[1] = pack2_bf16(a.z, a.w);
  o[2] = pack2_bf16(b.x, b.y);
  o[3] = pack2_bf16(b.z, b.w);
  *(v4u*)(out + i) = o;
}

// ---------------------------------------------------------------------------
// GRU cell with all GEMM operands pre-converted to bf16.
// grid = (H/16, B/16), block = 32 (one wave per 16x16 output tile).
// ---------------------------------------------------------------------------
__global__ __launch_bounds__(32) void gru_cell_kernel(
    const __bf16* __restrict__ x, int Kx,       // [B, Kx] bf16
    const __bf16* __restrict__ w_ih,            // [3H, Kx] bf16
    const __bf16* __restrict__ hA,              // [B, H] bf16 (GEMM operand)
    const float*  __restrict__ h32,             // [B, H] f32 (blend operand)
    const __bf16* __restrict__ w_hh,            // [3H, H] bf16
    const float* __restrict__ b_ih, const float* __restrict__ b_hh,
    float* __restrict__ out, __bf16* __restrict__ outbf) // [B, H]
{
  const int lane = threadIdx.x & 31;
  const int m16  = lane & 15;
  const int hi   = lane >> 4;
  const int nb   = blockIdx.x * 16;
  const int mb   = blockIdx.y * 16;

  v8f ir{}, iz{}, in_{}, hr{}, hz{}, hn{};

  { // input gates: x @ w_ih^T
    const __bf16* xrow = x    + (size_t)(mb + m16) * Kx + hi * 8;
    const __bf16* wr   = w_ih + (size_t)(nb + m16)         * Kx + hi * 8;
    const __bf16* wz   = w_ih + (size_t)(nb + m16 + H)     * Kx + hi * 8;
    const __bf16* wn   = w_ih + (size_t)(nb + m16 + 2 * H) * Kx + hi * 8;
    for (int k = 0; k < Kx; k += 32) {
      const v16bf a = load_a_bf16(xrow + k);
      ir  = wmma_bf16(a, load_b_bf16(wr + k), ir);
      iz  = wmma_bf16(a, load_b_bf16(wz + k), iz);
      in_ = wmma_bf16(a, load_b_bf16(wn + k), in_);
    }
  }
  { // hidden gates: h @ w_hh^T
    const __bf16* hrow = hA   + (size_t)(mb + m16) * H + hi * 8;
    const __bf16* wr   = w_hh + (size_t)(nb + m16)         * H + hi * 8;
    const __bf16* wz   = w_hh + (size_t)(nb + m16 + H)     * H + hi * 8;
    const __bf16* wn   = w_hh + (size_t)(nb + m16 + 2 * H) * H + hi * 8;
    for (int k = 0; k < H; k += 32) {
      const v16bf a = load_a_bf16(hrow + k);
      hr = wmma_bf16(a, load_b_bf16(wr + k), hr);
      hz = wmma_bf16(a, load_b_bf16(wz + k), hz);
      hn = wmma_bf16(a, load_b_bf16(wn + k), hn);
    }
  }

  const int ng = nb + m16;
  const float bir = b_ih[ng], biz = b_ih[ng + H], bin = b_ih[ng + 2 * H];
  const float bhr = b_hh[ng], bhz = b_hh[ng + H], bhn = b_hh[ng + 2 * H];
#pragma unroll
  for (int r = 0; r < 8; ++r) {
    const int m = mb + hi * 8 + r;
    const float rg = sigm(ir[r] + bir + hr[r] + bhr);
    const float zg = sigm(iz[r] + biz + hz[r] + bhz);
    const float nn = fast_tanh(in_[r] + bin + rg * (hn[r] + bhn));
    const float hv = (1.0f - zg) * nn + zg * h32[(size_t)m * H + ng];
    out[(size_t)m * H + ng] = hv;
    outbf[(size_t)m * H + ng] = (__bf16)hv;
  }
}

// ---------------------------------------------------------------------------
// q = hidden1 @ wq^T + bq (bf16 operands). grid = (H/16, B/16), block = 32.
// ---------------------------------------------------------------------------
__global__ __launch_bounds__(32) void qproj_kernel(
    const __bf16* __restrict__ h1, const __bf16* __restrict__ wq,
    const float* __restrict__ bq, float* __restrict__ q)
{
  const int lane = threadIdx.x & 31;
  const int m16  = lane & 15;
  const int hi   = lane >> 4;
  const int nb   = blockIdx.x * 16;
  const int mb   = blockIdx.y * 16;

  v8f acc{};
  const __bf16* arow = h1 + (size_t)(mb + m16) * H + hi * 8;
  const __bf16* brow = wq + (size_t)(nb + m16) * H + hi * 8;
  for (int k = 0; k < H; k += 32)
    acc = wmma_bf16(load_a_bf16(arow + k), load_b_bf16(brow + k), acc);

  const int ng = nb + m16;
  const float bb = bq[ng];
#pragma unroll
  for (int r = 0; r < 8; ++r)
    q[(size_t)(mb + hi * 8 + r) * H + ng] = acc[r] + bb;
}

// ---------------------------------------------------------------------------
extern "C" void kernel_launch(void* const* d_in, const int* in_sizes, int n_in,
                              void* d_out, int out_size, void* d_ws, size_t ws_size,
                              hipStream_t stream) {
  const float* input   = (const float*)d_in[0];
  const float* hidden  = (const float*)d_in[1];
  const float* context = (const float*)d_in[2];
  const float* w_ih1   = (const float*)d_in[3];
  const float* w_hh1   = (const float*)d_in[4];
  const float* b_ih1   = (const float*)d_in[5];
  const float* b_hh1   = (const float*)d_in[6];
  const float* wq      = (const float*)d_in[7];
  const float* bq      = (const float*)d_in[8];
  const float* wk      = (const float*)d_in[9];
  const float* bk      = (const float*)d_in[10];
  const float* wv      = (const float*)d_in[11];
  const float* bv      = (const float*)d_in[12];
  const float* w_ih2   = (const float*)d_in[13];
  const float* w_hh2   = (const float*)d_in[14];
  const float* b_ih2   = (const float*)d_in[15];
  const float* b_hh2   = (const float*)d_in[16];

  float* out      = (float*)d_out;
  float* h2_out   = out;                      // [B, H]
  float* attn_out = out + (size_t)B * H;      // [B, CTX]

  // Workspace layout: f32 activations first, then bf16 staging region.
  float*  h1  = (float*)d_ws;                 // [B, H]
  float*  q   = h1 + (size_t)B * H;           // [B, H]
  __bf16* xbf     = (__bf16*)(q + (size_t)B * H);
  __bf16* hbf     = xbf    + (size_t)B * IN;
  __bf16* h1bf    = hbf    + (size_t)B * H;
  __bf16* attnbf  = h1bf   + (size_t)B * H;
  __bf16* wih1bf  = attnbf + (size_t)B * CTX;
  __bf16* whh1bf  = wih1bf + (size_t)3 * H * IN;
  __bf16* wqbf    = whh1bf + (size_t)3 * H * H;
  __bf16* wkbf    = wqbf   + (size_t)H * H;
  __bf16* wih2bf  = wkbf   + (size_t)H * CTX;
  __bf16* whh2bf  = wih2bf + (size_t)3 * H * CTX;
  __bf16* h2bf    = xbf;  // reuse: input bf16 is dead by the final GRU

  // 0) One-time f32 -> bf16 staging of all reused GEMM operands.
  auto cvt = [&](const float* src, __bf16* dst, size_t n) {
    cvt_kernel<<<dim3((unsigned)(n / 2048)), 256, 0, stream>>>(src, dst,
                                                               (unsigned)n);
  };
  cvt(input,  xbf,    (size_t)B * IN);
  cvt(hidden, hbf,    (size_t)B * H);
  cvt(w_ih1,  wih1bf, (size_t)3 * H * IN);
  cvt(w_hh1,  whh1bf, (size_t)3 * H * H);
  cvt(wq,     wqbf,   (size_t)H * H);
  cvt(wk,     wkbf,   (size_t)H * CTX);
  cvt(w_ih2,  wih2bf, (size_t)3 * H * CTX);
  cvt(w_hh2,  whh2bf, (size_t)3 * H * H);

  const dim3 gemm_grid(H / 16, B / 16); // (64, 16) tiles, 1 wave each

  // 1) GRU cell 1 -> hidden1 (f32 + bf16 copies)
  gru_cell_kernel<<<gemm_grid, 32, 0, stream>>>(
      xbf, IN, wih1bf, hbf, hidden, whh1bf, b_ih1, b_hh1, h1, h1bf);
  // 2) q = hidden1 @ wq^T + bq
  qproj_kernel<<<gemm_grid, 32, 0, stream>>>(h1bf, wqbf, bq, q);
  // 3) fused Bahdanau attention -> attn_values (f32 into d_out, bf16 into ws)
  attn_kernel<<<dim3(B), 256, 0, stream>>>(context, wkbf, bk, q, wv, bv,
                                           attn_out, attnbf);
  // 4) GRU cell 2 -> hidden2
  gru_cell_kernel<<<gemm_grid, 32, 0, stream>>>(
      attnbf, CTX, wih2bf, h1bf, h1, whh2bf, b_ih2, b_hh2, h2_out, h2bf);
}
